// NormalizedAttention_81003083202669
// MI455X (gfx1250) — matrix operand order repaired and checked by reading, not compiled, not run on hardware
//
#include <hip/hip_runtime.h>

// Problem constants (match reference): B=2, N=2048, DIM=1024, H=16, D=64
#define BB   2
#define NN   2048
#define DIMC 1024
#define HH   16
#define DH   64

typedef __attribute__((ext_vector_type(16))) __bf16 v16bf;
typedef __attribute__((ext_vector_type(8)))  __bf16 v8bf;
typedef __attribute__((ext_vector_type(8)))  float  v8f;

__device__ __forceinline__ v8f zero8() {
  v8f z = {0.f, 0.f, 0.f, 0.f, 0.f, 0.f, 0.f, 0.f};
  return z;
}

__device__ __forceinline__ v8f wmma_bf16(v16bf a, v16bf b, v8f c) {
  // D = A(16xK) * B(Kx16) + C, K=32, bf16 in / f32 accumulate
  return __builtin_amdgcn_wmma_f32_16x16x32_bf16(false, a, false, b, (short)0, c,
                                                 false, false);
}

// ---- WMMA operand loaders (wave32, contiguous k order, same for A and B) -------------------
// A operand, logical 16x32 (M x K) tile, row-major source with leading dim ld.
// lanes 0-15: row M=lane, K k0..k0+15
// lanes 16-31: row M=lane-16, K k0+16..k0+31
__device__ __forceinline__ v16bf load_a(const __bf16* base, int ld, int row0,
                                        int k0, int lane) {
  int r   = row0 + (lane & 15);
  int sel = lane >> 4;
  const __bf16* p = base + (size_t)r * ld + k0 + sel * 16;
  v8bf lo = *(const v8bf*)p;
  v8bf hi = *(const v8bf*)(p + 8);
  return __builtin_shufflevector(lo, hi, 0, 1, 2, 3, 4, 5, 6, 7,
                                 8, 9, 10, 11, 12, 13, 14, 15);
}

// B operand, logical 32x16 (K x N) tile; source is B^T stored row-major
// (N-major, leading dim ld). lane = N column; lanes 0-15 hold K k0..k0+15,
// lanes 16-31 hold K k0+16..k0+31 -> one contiguous 32B load.
__device__ __forceinline__ v16bf load_b(const __bf16* baseT, int ld, int col0,
                                        int k0, int lane) {
  int c   = col0 + (lane & 15);
  int sel = lane >> 4;
  return *(const v16bf*)(baseT + (size_t)c * ld + k0 + sel * 16);
}

// ---- Prep kernels ----------------------------------------------------------
__global__ __launch_bounds__(256) void cvt_f32_to_bf16(const float* __restrict__ in,
                                                       __bf16* __restrict__ out,
                                                       long n) {
  long i = (long)blockIdx.x * blockDim.x + threadIdx.x;
  if (i < n) out[i] = (__bf16)in[i];
}

// WT[n][k] = (bf16) W[k][n]   (W is Krows x Ncols row-major)
__global__ __launch_bounds__(256) void transpose_cvt(const float* __restrict__ W,
                                                     __bf16* __restrict__ WT,
                                                     int Krows, int Ncols) {
  long i = (long)blockIdx.x * blockDim.x + threadIdx.x;
  if (i >= (long)Krows * Ncols) return;
  int k = (int)(i % Krows);
  int n = (int)(i / Krows);
  WT[(size_t)n * Krows + k] = (__bf16)W[(size_t)k * Ncols + n];
}

// ---- QKV projection: qkv = x @ Wqkv + b, scattered into Q,K (B,H,N,D) and
// ---- V^T (B,H,D,N), Q pre-scaled by 1/sqrt(D). One wave = 16x64 out strip.
// ---- Ping-pong pipelined k loop: loads write directly into the buffer used
// ---- one stage later -> no register-rotation copies.
__global__ __launch_bounds__(256) void qkv_gemm(const __bf16* __restrict__ xb,
                                                const __bf16* __restrict__ WT,
                                                const float* __restrict__ bias,
                                                __bf16* __restrict__ Q,
                                                __bf16* __restrict__ K,
                                                __bf16* __restrict__ Vt) {
  const int lane = threadIdx.x & 31;
  const int wid  = (int)((blockIdx.x * blockDim.x + threadIdx.x) >> 5);
  const int NS   = (3 * DIMC) / 64;  // 48 column strips
  int ns = wid % NS, mt = wid / NS;
  if (mt >= (BB * NN) / 16) return;
  int m0 = mt * 16, n0 = ns * 64;

  v8f acc[4];
#pragma unroll
  for (int t = 0; t < 4; ++t) acc[t] = zero8();

  v16bf aA, bA[4];
  aA = load_a(xb, DIMC, m0, 0, lane);
#pragma unroll
  for (int t = 0; t < 4; ++t) bA[t] = load_b(WT, DIMC, n0 + t * 16, 0, lane);

  for (int k = 0; k < DIMC; k += 64) {
    // stage B operands (k+32) fly over stage-A WMMAs
    v16bf aB = load_a(xb, DIMC, m0, k + 32, lane);
    v16bf bB[4];
#pragma unroll
    for (int t = 0; t < 4; ++t) bB[t] = load_b(WT, DIMC, n0 + t * 16, k + 32, lane);
    __builtin_prefetch(WT + (size_t)(n0 + (lane & 15)) * DIMC +
                           ((k + 128) & (DIMC - 1)), 0, 1);
#pragma unroll
    for (int t = 0; t < 4; ++t) acc[t] = wmma_bf16(aA, bA[t], acc[t]);

    // refill stage A (k+64, wrapped on last iter; result discarded)
    int k2 = (k + 64) & (DIMC - 1);
    aA = load_a(xb, DIMC, m0, k2, lane);
#pragma unroll
    for (int t = 0; t < 4; ++t) bA[t] = load_b(WT, DIMC, n0 + t * 16, k2, lane);
#pragma unroll
    for (int t = 0; t < 4; ++t) acc[t] = wmma_bf16(aB, bB[t], acc[t]);
  }

  // C layout: lane = output col, component i -> row m0 + i + 8*(lane>=16)
  int sel8 = (lane >> 4) * 8;
#pragma unroll
  for (int t = 0; t < 4; ++t) {
    int c = n0 + t * 16 + (lane & 15);
    float bv  = bias[c];
    int three = c >> 10;          // 0:Q 1:K 2:V
    int h     = (c & 1023) >> 6;
    int d     = c & 63;
#pragma unroll
    for (int i = 0; i < 8; ++i) {
      int tok = m0 + sel8 + i;
      int b_  = tok >> 11;        // tok / NN
      int n   = tok & (NN - 1);
      float v = acc[t][i] + bv;
      size_t bh = (size_t)(b_ * HH + h);
      if (three == 0)      Q[(bh * NN + n) * DH + d] = (__bf16)(v * 0.125f);
      else if (three == 1) K[(bh * NN + n) * DH + d] = (__bf16)v;
      else                 Vt[(bh * DH + d) * NN + n] = (__bf16)v;
    }
  }
}

// ---- Fused normalized attention. One wave owns 16 queries of one (b,h).
// Phase 1: S^T = K @ Q^T tiles (WMMA), accumulate sum(s^2) per query row.
// Phase 2: recompute S^T, p = exp(gamma*s/denom), accumulate O^T = V^T @ P^T.
__global__ __launch_bounds__(256) void attn_fused(const __bf16* __restrict__ Q,
                                                  const __bf16* __restrict__ K,
                                                  const __bf16* __restrict__ Vt,
                                                  const float* __restrict__ sigmaP,
                                                  const float* __restrict__ gammaP,
                                                  __bf16* __restrict__ out) {
  const int lane = threadIdx.x & 31;
  const int wid  = (int)((blockIdx.x * blockDim.x + threadIdx.x) >> 5);
  const int QT   = NN / 16;  // 128 query tiles per head
  int qt = wid % QT, bh = wid / QT;
  if (bh >= BB * HH) return;

  const __bf16* Qp = Q  + (size_t)bh * NN * DH;
  const __bf16* Kp = K  + (size_t)bh * NN * DH;
  const __bf16* Vp = Vt + (size_t)bh * DH * NN;
  int q0 = qt * 16;

  // Q^T B-operands (K dim = head_dim 64, two 32-slices); resident all kernel.
  v16bf qb0 = load_b(Qp, DH, q0, 0, lane);
  v16bf qb1 = load_b(Qp, DH, q0, 32, lane);

  // 4 A-operands covering a 32-key group: {keys n0..n0+15, n0+16..n0+31} x
  // {head-dim 0..31, 32..63}
  auto load_k4 = [&](v16bf (&ka)[4], int n0) {
    ka[0] = load_a(Kp, DH, n0, 0, lane);
    ka[1] = load_a(Kp, DH, n0, 32, lane);
    ka[2] = load_a(Kp, DH, n0 + 16, 0, lane);
    ka[3] = load_a(Kp, DH, n0 + 16, 32, lane);
  };

  // ---- Phase 1: sum of squared (pre-scaled) scores over all keys ----------
  float sumsq = 0.f;
  auto sq_step = [&](const v16bf (&ka)[4]) {
    v8f c0 = zero8();
    c0 = wmma_bf16(ka[0], qb0, c0); c0 = wmma_bf16(ka[1], qb1, c0);
    v8f c1 = zero8();
    c1 = wmma_bf16(ka[2], qb0, c1); c1 = wmma_bf16(ka[3], qb1, c1);
#pragma unroll
    for (int i = 0; i < 8; ++i) sumsq += c0[i] * c0[i] + c1[i] * c1[i];
  };
  {
    v16bf kaA[4], kaB[4];
    load_k4(kaA, 0);
    for (int n0 = 0; n0 < NN; n0 += 64) {
      load_k4(kaB, n0 + 32);             // fly over stage-A WMMAs
      sq_step(kaA);
      load_k4(kaA, (n0 + 64) & (NN - 1));  // refill A (wrap discarded)
      sq_step(kaB);
    }
  }
  // lanes l and l+16 hold the same query column -> half-wave reduce
  sumsq += __shfl_xor(sumsq, 16, 32);
  float denom = sigmaP[0] + sqrtf(sumsq * (1.0f / (float)NN));
  float coef  = gammaP[0] / denom;  // |coef*s| <= gamma*sqrt(N): exp is safe

  // ---- Phase 2: softmax numerators + O^T = V^T @ P^T ----------------------
  v8f o[4];
#pragma unroll
  for (int t = 0; t < 4; ++t) o[t] = zero8();
  float ssum = 0.f;
  bool hi = (lane >> 4) != 0;

  auto pv_step = [&](const v16bf (&ka)[4], int n0) {
    v8f c0 = zero8();
    c0 = wmma_bf16(ka[0], qb0, c0); c0 = wmma_bf16(ka[1], qb1, c0);
    v8f c1 = zero8();
    c1 = wmma_bf16(ka[2], qb0, c1); c1 = wmma_bf16(ka[3], qb1, c1);

    // V tiles issued before the exp/pack VALU block: transcendental work
    // covers their latency before the PV WMMAs.
    v16bf va[4];
#pragma unroll
    for (int t = 0; t < 4; ++t) va[t] = load_a(Vp, NN, t * 16, n0, lane);

    float e0[8], e1[8];
#pragma unroll
    for (int i = 0; i < 8; ++i) {
      e0[i] = __expf(coef * c0[i]);
      e1[i] = __expf(coef * c1[i]);
      ssum += e0[i] + e1[i];
    }
    // Pack P as B operand of V^T @ P^T: lane = query col; lanes<16 need key
    // rows n0..n0+15, lanes>=16 need n0+16..n0+31. Half of each comes from
    // the partner lane (xor 16) -- single cross-lane exchange, no LDS.
    v16bf pb;
#pragma unroll
    for (int i = 0; i < 8; ++i) {
      float p0 = __shfl_xor(e0[i], 16, 32);
      float p1 = __shfl_xor(e1[i], 16, 32);
      pb[i]     = (__bf16)(hi ? p1 : e0[i]);
      pb[i + 8] = (__bf16)(hi ? e1[i] : p0);
    }
#pragma unroll
    for (int t = 0; t < 4; ++t) o[t] = wmma_bf16(va[t], pb, o[t]);
  };
  {
    v16bf kaA[4], kaB[4];
    load_k4(kaA, 0);
    for (int n0 = 0; n0 < NN; n0 += 64) {
      load_k4(kaB, n0 + 32);
      pv_step(kaA, n0);
      load_k4(kaA, (n0 + 64) & (NN - 1));
      pv_step(kaB, n0 + 32);
    }
  }
  ssum += __shfl_xor(ssum, 16, 32);
  float inv = 1.0f / ssum;

  // O^T C layout: lane = query col, comp i -> d = t*16 + i + 8*(lane>=16)
  int b_ = bh / HH, h = bh % HH;
  int q  = q0 + (lane & 15);
  int sel8 = (lane >> 4) * 8;
  __bf16* orow = out + ((size_t)(b_ * NN + q)) * DIMC + h * DH;
#pragma unroll
  for (int t = 0; t < 4; ++t) {
#pragma unroll
    for (int i = 0; i < 8; ++i) {
      orow[t * 16 + sel8 + i] = (__bf16)(o[t][i] * inv);
    }
  }
}

// ---- Output projection: out = attn_out @ Wproj + b (fp32 result) -----------
__global__ __launch_bounds__(256) void proj_gemm(const __bf16* __restrict__ ab,
                                                 const __bf16* __restrict__ WT,
                                                 const float* __restrict__ bias,
                                                 float* __restrict__ out) {
  const int lane = threadIdx.x & 31;
  const int wid  = (int)((blockIdx.x * blockDim.x + threadIdx.x) >> 5);
  const int NS   = DIMC / 64;  // 16 column strips
  int ns = wid % NS, mt = wid / NS;
  if (mt >= (BB * NN) / 16) return;
  int m0 = mt * 16, n0 = ns * 64;

  v8f acc[4];
#pragma unroll
  for (int t = 0; t < 4; ++t) acc[t] = zero8();

  v16bf aA, bA[4];
  aA = load_a(ab, DIMC, m0, 0, lane);
#pragma unroll
  for (int t = 0; t < 4; ++t) bA[t] = load_b(WT, DIMC, n0 + t * 16, 0, lane);

  for (int k = 0; k < DIMC; k += 64) {
    v16bf aB = load_a(ab, DIMC, m0, k + 32, lane);
    v16bf bB[4];
#pragma unroll
    for (int t = 0; t < 4; ++t) bB[t] = load_b(WT, DIMC, n0 + t * 16, k + 32, lane);
    __builtin_prefetch(WT + (size_t)(n0 + (lane & 15)) * DIMC +
                           ((k + 128) & (DIMC - 1)), 0, 1);
#pragma unroll
    for (int t = 0; t < 4; ++t) acc[t] = wmma_bf16(aA, bA[t], acc[t]);

    int k2 = (k + 64) & (DIMC - 1);
    aA = load_a(ab, DIMC, m0, k2, lane);
#pragma unroll
    for (int t = 0; t < 4; ++t) bA[t] = load_b(WT, DIMC, n0 + t * 16, k2, lane);
#pragma unroll
    for (int t = 0; t < 4; ++t) acc[t] = wmma_bf16(aB, bB[t], acc[t]);
  }

  int sel8 = (lane >> 4) * 8;
#pragma unroll
  for (int t = 0; t < 4; ++t) {
    int c = n0 + t * 16 + (lane & 15);
    float bv = bias[c];
#pragma unroll
    for (int i = 0; i < 8; ++i) {
      int tok = m0 + sel8 + i;
      out[(size_t)tok * DIMC + c] = acc[t][i] + bv;
    }
  }
}

extern "C" void kernel_launch(void* const* d_in, const int* in_sizes, int n_in,
                              void* d_out, int out_size, void* d_ws, size_t ws_size,
                              hipStream_t stream) {
  const float* x     = (const float*)d_in[0];
  const float* Wqkv  = (const float*)d_in[1];
  const float* bqkv  = (const float*)d_in[2];
  const float* Wproj = (const float*)d_in[3];
  const float* bproj = (const float*)d_in[4];
  const float* sigma = (const float*)d_in[5];
  const float* gamma = (const float*)d_in[6];
  float* out = (float*)d_out;

  // Workspace carve (bf16, 256B aligned): ~48 MB total
  char* w = (char*)d_ws;
  size_t off = 0;
  auto carve = [&](size_t elems) {
    void* p = w + off;
    off += (elems * sizeof(__bf16) + 255) & ~(size_t)255;
    return p;
  };
  __bf16* xb  = (__bf16*)carve((size_t)BB * NN * DIMC);      // x in bf16
  __bf16* WqT = (__bf16*)carve((size_t)3 * DIMC * DIMC);     // Wqkv^T bf16
  __bf16* WpT = (__bf16*)carve((size_t)DIMC * DIMC);         // Wproj^T bf16
  __bf16* Qw  = (__bf16*)carve((size_t)BB * HH * NN * DH);   // Q (pre-scaled)
  __bf16* Kw  = (__bf16*)carve((size_t)BB * HH * NN * DH);   // K
  __bf16* Vtw = (__bf16*)carve((size_t)BB * HH * DH * NN);   // V^T
  __bf16* Ab  = (__bf16*)carve((size_t)BB * NN * DIMC);      // attention out

  // Prep: convert / transpose weights once per call
  cvt_f32_to_bf16<<<(BB * NN * DIMC) / 256, 256, 0, stream>>>(x, xb,
                                                              (long)BB * NN * DIMC);
  transpose_cvt<<<(3 * DIMC * DIMC) / 256, 256, 0, stream>>>(Wqkv, WqT, DIMC,
                                                             3 * DIMC);
  transpose_cvt<<<(DIMC * DIMC) / 256, 256, 0, stream>>>(Wproj, WpT, DIMC, DIMC);

  // QKV projection: 256 M-tiles x 48 strips = 12288 waves, 8 waves/block
  qkv_gemm<<<(256 * 48) / 8, 256, 0, stream>>>(xb, WqT, bqkv, Qw, Kw, Vtw);

  // Fused attention: B*H*(N/16) = 4096 waves
  attn_fused<<<(BB * HH * (NN / 16)) / 8, 256, 0, stream>>>(Qw, Kw, Vtw, sigma,
                                                            gamma, Ab);

  // Output projection: 256 x 16 = 4096 waves
  proj_gemm<<<(256 * 16) / 8, 256, 0, stream>>>(Ab, WpT, bproj, out);
}
